// LaneATT_86182813761557
// MI455X (gfx1250) — matrix operand order repaired and checked by reading, not compiled, not run on hardware
//
#include <hip/hip_runtime.h>
#include <hip/hip_bf16.h>
#include <stdint.h>

// ---- problem constants ----
#define BB    32
#define NN    1000
#define CC    64
#define HHF   12
#define WWF   20
#define DD    768          // C*HF
#define NM1   999          // N-1 attention columns
#define NPM   1024         // padded row count (N -> 1024)
#define NPC   1024         // padded logit/attention columns (999 -> 1024)
#define HCOL  80           // padded head columns (2+73=75 -> 80)
#define KHEAD 1536         // 2*D
#define NOUT  77           // output channels per (b,n)

#define LDS_STRIDE 40      // bf16 elems per LDS tile row (32 + 8 pad; 80B, 16B-aligned)

typedef __attribute__((ext_vector_type(16))) __bf16 bf16x16;
typedef __attribute__((ext_vector_type(8)))  __bf16 bf16x8;
typedef __attribute__((ext_vector_type(8)))  float  f32x8;
typedef __attribute__((ext_vector_type(4)))  int    v4i;

union BFrag { bf16x16 v; bf16x8 h[2]; };

#ifndef __has_builtin
#define __has_builtin(x) 0
#endif

#if __has_builtin(__builtin_amdgcn_global_load_async_to_lds_b128)
#define HAVE_ASYNC_LDS 1
typedef __attribute__((address_space(1))) v4i* as1_v4i_p;   // global int4*
typedef __attribute__((address_space(3))) v4i* as3_v4i_p;   // LDS int4*
#else
#define HAVE_ASYNC_LDS 0
#endif

#if __has_builtin(__builtin_amdgcn_s_wait_asynccnt)
#define WAIT_ASYNC(n) __builtin_amdgcn_s_wait_asynccnt(n)
#else
#define WAIT_ASYNC(n) asm volatile("s_wait_asynccnt %0" ::"i"(n) : "memory")
#endif

#define WMMA_BF16(a, b, c) \
  __builtin_amdgcn_wmma_f32_16x16x32_bf16(false, (a), false, (b), (short)0, (c), false, false)

// A-matrix fragment (16x32 bf16, M x K), A row-major with leading dim lda.
// Lane L -> row (L&15); this code's contiguous K run [16*(L>>4), +16), matching B.
__device__ __forceinline__ bf16x16 load_a_frag(const __bf16* A, int lda, int m0, int k0, int lane) {
  int row  = m0 + (lane & 15);
  int koff = k0 + ((lane >> 4) << 4);
  const __bf16* p = A + (size_t)row * lda + koff;
  BFrag u;
  u.h[0] = *(const bf16x8*)(p);
  u.h[1] = *(const bf16x8*)(p + 8);
  return u.v;
}

// B-matrix fragment (32x16 bf16, K x N) from an LDS tile holding 64 columns
// of 32 K-values each (row-major, stride LDS_STRIDE). Lane L -> column
// 16*t + (L&15); contiguous K run [16*(L>>4), +16).
__device__ __forceinline__ bf16x16 load_b_frag_lds(const __bf16* tile, int t, int lane) {
  int c    = 16 * t + (lane & 15);
  int koff = (lane >> 4) << 4;
  const __bf16* p = tile + c * LDS_STRIDE + koff;
  BFrag u;
  u.h[0] = *(const bf16x8*)(p);
  u.h[1] = *(const bf16x8*)(p + 8);
  return u.v;
}

// B-matrix fragment straight from global (used by the small head GEMM).
__device__ __forceinline__ bf16x16 load_b_frag(const __bf16* Bt, int ldb, int n0, int k0, int lane) {
  int col  = n0 + (lane & 15);
  int koff = k0 + ((lane >> 4) << 4);
  const __bf16* p = Bt + (size_t)col * ldb + koff;
  BFrag u;
  u.h[0] = *(const bf16x8*)(p);
  u.h[1] = *(const bf16x8*)(p + 8);
  return u.v;
}

// Copy one 64x32 bf16 B tile (cols n0..n0+63, K slice [k,k+32)) into LDS.
// 256 x 16B chunks; 128 threads issue 2 async b128 ops each.
__device__ __forceinline__ void stage_copy(__bf16* dst, const __bf16* Bt, int ldb,
                                           int n0, int k, int tid) {
#pragma unroll
  for (int j = 0; j < 2; ++j) {
    int q = tid + 128 * j;        // 0..255
    int c = q >> 2;               // column 0..63
    int p = q & 3;                // 16B chunk within the 64B row
    const __bf16* src = Bt + (size_t)(n0 + c) * ldb + k + 8 * p;
    __bf16* d = dst + c * LDS_STRIDE + 8 * p;
#if HAVE_ASYNC_LDS
    __builtin_amdgcn_global_load_async_to_lds_b128((as1_v4i_p)src, (as3_v4i_p)d, 0, 0);
#else
    *(bf16x8*)d = *(const bf16x8*)src;
#endif
  }
}

// One 32-deep K step: two A fragments (rows m0, m0+16) against the staged
// 64-column B tile -> 8 WMMAs; each B fragment ds_load feeds 2 WMMAs.
template <int LDA>
__device__ __forceinline__ void compute_step(const __bf16* A, const __bf16* cur,
                                             int m0, int k, int lane, f32x8 acc[2][4]) {
  __builtin_prefetch(A + (size_t)(m0 + (lane & 15)) * LDA + k + 64, 0, 1);
  bf16x16 a0 = load_a_frag(A, LDA, m0,      k, lane);
  bf16x16 a1 = load_a_frag(A, LDA, m0 + 16, k, lane);
#pragma unroll
  for (int t = 0; t < 4; ++t) {
    bf16x16 bf = load_b_frag_lds(cur, t, lane);
    acc[0][t] = WMMA_BF16(a0, bf, acc[0][t]);
    acc[1][t] = WMMA_BF16(a1, bf, acc[1][t]);
  }
}

// 128x64 block GEMM: 4 waves, wave w computes 32 rows (m0 = base + 32w).
// B tile double-buffered in LDS via async copies, branch-free unroll-by-2
// steady state with a peeled 2-stage tail. Invariant at loop head: lds0
// holds stage s, and at most its own 2 async ops are outstanding.
template <int KTOT, int LDA, int LDB>
__device__ __forceinline__ void gemm_block_128x64(
    const __bf16* A, const __bf16* Bt, __bf16* lds0, __bf16* lds1,
    int m0, int n0, int lane, int tid, f32x8 acc[2][4]) {
  constexpr int NK = KTOT / 32;
  static_assert(NK >= 4 && (NK % 2) == 0, "pipeline needs even NK >= 4");
  stage_copy(lds0, Bt, LDB, n0, 0, tid);
#pragma unroll 1
  for (int s = 0; s + 3 < NK; s += 2) {
    stage_copy(lds1, Bt, LDB, n0, (s + 1) * 32, tid);
    WAIT_ASYNC(2);                       // stage s complete (in-order)
    __syncthreads();
    compute_step<LDA>(A, lds0, m0, s * 32, lane, acc);
    __syncthreads();                     // lds0 free for refill
    stage_copy(lds0, Bt, LDB, n0, (s + 2) * 32, tid);
    WAIT_ASYNC(2);                       // stage s+1 complete
    __syncthreads();
    compute_step<LDA>(A, lds1, m0, (s + 1) * 32, lane, acc);
    __syncthreads();                     // lds1 free for refill
  }
  // tail: lds0 holds stage NK-2; stage NK-1 into lds1
  stage_copy(lds1, Bt, LDB, n0, (NK - 1) * 32, tid);
  WAIT_ASYNC(2);
  __syncthreads();
  compute_step<LDA>(A, lds0, m0, (NK - 2) * 32, lane, acc);
  WAIT_ASYNC(0);
  __syncthreads();
  compute_step<LDA>(A, lds1, m0, (NK - 1) * 32, lane, acc);
}

// ---------------------------------------------------------------------------
// Kernel 1: gather ROI features, mask, convert to bf16; write feat (B,NPM,DD)
// row-major AND featT (B,DD,NPM) so feat can be a WMMA B-operand later.
// ---------------------------------------------------------------------------
__global__ __launch_bounds__(256) void lane_gather_kernel(
    const float* __restrict__ FV, const int* __restrict__ x_idx,
    const uint8_t* __restrict__ inval, __bf16* __restrict__ feat,
    __bf16* __restrict__ featT) {
  int n = blockIdx.x;   // 0..NPM-1
  int b = blockIdx.y;   // 0..BB-1
  for (int d = threadIdx.x; d < DD; d += 256) {
    __bf16 v = (__bf16)0.0f;
    if (n < NN) {
      int c = d / HHF, h = d % HHF;
      int x = x_idx[(size_t)n * (CC * HHF) + h];          // c-broadcast: c=0 slice
      bool bad = inval[(size_t)n * (CC * HHF) + h] != 0;  // c-broadcast mask
      float f = bad ? 0.0f : FV[(((size_t)b * CC + c) * HHF + h) * WWF + x];
      v = (__bf16)f;
    }
    feat [((size_t)b * NPM + n) * DD + d]  = v;
    featT[((size_t)b * DD  + d) * NPM + n] = v;
  }
}

// ---------------------------------------------------------------------------
// Kernel 2: attn_w (768 x 999) -> Wt1 (NPC x 768) bf16 transposed
// ---------------------------------------------------------------------------
__global__ __launch_bounds__(256) void lane_build_wt1_kernel(
    const float* __restrict__ aw, __bf16* __restrict__ Wt) {
  int j = blockIdx.x;  // 0..NPC-1
  for (int k = threadIdx.x; k < DD; k += 256) {
    float v = (j < NM1) ? aw[(size_t)k * NM1 + j] : 0.0f;
    Wt[(size_t)j * DD + k] = (__bf16)v;
  }
}

// ---------------------------------------------------------------------------
// Kernel 3: [cls_w (1536x2) | reg_w (1536x73)] -> hW (HCOL x 1536) bf16 transposed
// ---------------------------------------------------------------------------
__global__ __launch_bounds__(256) void lane_build_hw_kernel(
    const float* __restrict__ cw, const float* __restrict__ rw,
    __bf16* __restrict__ W) {
  int j = blockIdx.x;  // 0..HCOL-1
  for (int k = threadIdx.x; k < KHEAD; k += 256) {
    float v = 0.0f;
    if (j < 2)       v = cw[(size_t)k * 2 + j];
    else if (j < 75) v = rw[(size_t)k * 73 + (j - 2)];
    W[(size_t)j * KHEAD + k] = (__bf16)v;
  }
}

// ---------------------------------------------------------------------------
// Kernel 4 (GEMM1): logits = feat @ attn_w + attn_b   (M=NPM, N=NPC, K=DD)
// ---------------------------------------------------------------------------
__global__ __launch_bounds__(128) void lane_gemm1_kernel(
    const __bf16* __restrict__ feat, const __bf16* __restrict__ Wt,
    const float* __restrict__ attn_b, float* __restrict__ logits) {
  __shared__ __bf16 ldsB[2][64 * LDS_STRIDE];
  int b = blockIdx.z;
  int tid = threadIdx.x, wave = tid >> 5, lane = tid & 31;
  int m0 = blockIdx.x * 128 + wave * 32;
  int n0 = blockIdx.y * 64;
  const __bf16* A = feat + (size_t)b * NPM * DD;
  f32x8 acc[2][4] = {};
  gemm_block_128x64<DD, DD, DD>(A, Wt, ldsB[0], ldsB[1], m0, n0, lane, tid, acc);
  float* out = logits + (size_t)b * NPM * NPC;
#pragma unroll
  for (int mi = 0; mi < 2; ++mi)
#pragma unroll
    for (int t = 0; t < 4; ++t)
#pragma unroll
      for (int r = 0; r < 8; ++r) {
        int row = m0 + 16 * mi + r + ((lane >> 4) << 3);
        int col = n0 + 16 * t + (lane & 15);
        float v = acc[mi][t][r];
        v = (col < NM1) ? (v + attn_b[col]) : -1e30f;
        out[(size_t)row * NPC + col] = v;
      }
}

// ---------------------------------------------------------------------------
// Kernel 5: row softmax over 999 logits, scattered into S with the
// diagonal-skip column mapping (jj = j + (j >= i)); padded rows/cols zeroed.
// ---------------------------------------------------------------------------
__global__ __launch_bounds__(256) void lane_softmax_kernel(
    const float* __restrict__ logits, __bf16* __restrict__ S) {
  int i = blockIdx.x;  // 0..NPM-1
  int b = blockIdx.y;
  __bf16* srow = S + ((size_t)b * NPM + i) * NPC;
  int tid = threadIdx.x;
  if (i >= NN) {
    for (int j = tid; j < NPC; j += 256) srow[j] = (__bf16)0.0f;
    return;
  }
  const float* row = logits + ((size_t)b * NPM + i) * NPC;
  __shared__ float red[256];
  float mx = -1e30f;
  for (int j = tid; j < NM1; j += 256) mx = fmaxf(mx, row[j]);
  red[tid] = mx; __syncthreads();
  for (int s = 128; s > 0; s >>= 1) {
    if (tid < s) red[tid] = fmaxf(red[tid], red[tid + s]);
    __syncthreads();
  }
  mx = red[0]; __syncthreads();
  float sum = 0.0f;
  for (int j = tid; j < NM1; j += 256) sum += __expf(row[j] - mx);
  red[tid] = sum; __syncthreads();
  for (int s = 128; s > 0; s >>= 1) {
    if (tid < s) red[tid] += red[tid + s];
    __syncthreads();
  }
  float inv = 1.0f / red[0];
  for (int jj = tid; jj < NPC; jj += 256) {
    float v = 0.0f;
    if (jj != i && jj < NN) {
      int j = jj - (jj > i ? 1 : 0);  // j in [0, 998]
      v = __expf(row[j] - mx) * inv;
    }
    srow[jj] = (__bf16)v;
  }
}

// ---------------------------------------------------------------------------
// Kernel 6 (GEMM2): att_feat = S @ feat   (M=NPM, N=DD, K=NPC), bf16 output.
// ---------------------------------------------------------------------------
__global__ __launch_bounds__(128) void lane_gemm2_kernel(
    const __bf16* __restrict__ S, const __bf16* __restrict__ featT,
    __bf16* __restrict__ att) {
  __shared__ __bf16 ldsB[2][64 * LDS_STRIDE];
  int b = blockIdx.z;
  int tid = threadIdx.x, wave = tid >> 5, lane = tid & 31;
  int m0 = blockIdx.x * 128 + wave * 32;
  int n0 = blockIdx.y * 64;
  const __bf16* A  = S     + (size_t)b * NPM * NPC;
  const __bf16* Bt = featT + (size_t)b * DD * NPM;
  f32x8 acc[2][4] = {};
  gemm_block_128x64<NPC, NPC, NPM>(A, Bt, ldsB[0], ldsB[1], m0, n0, lane, tid, acc);
  __bf16* out = att + (size_t)b * NPM * DD;
#pragma unroll
  for (int mi = 0; mi < 2; ++mi)
#pragma unroll
    for (int t = 0; t < 4; ++t)
#pragma unroll
      for (int r = 0; r < 8; ++r) {
        int row = m0 + 16 * mi + r + ((lane >> 4) << 3);
        int col = n0 + 16 * t + (lane & 15);
        out[(size_t)row * DD + col] = (__bf16)acc[mi][t][r];
      }
}

// ---------------------------------------------------------------------------
// Kernel 7 (GEMM3 + epilogue): hcat = [att_feat | feat] @ hW (K=1536, N=HCOL).
// Concat realized by switching the A base pointer across the K loop.
// ---------------------------------------------------------------------------
__global__ __launch_bounds__(128) void lane_gemm3_kernel(
    const __bf16* __restrict__ att, const __bf16* __restrict__ feat,
    const __bf16* __restrict__ W, const float* __restrict__ cls_b,
    const float* __restrict__ reg_b, const float* __restrict__ anchors,
    float* __restrict__ out) {
  int b = blockIdx.y;
  int wave = threadIdx.x >> 5, lane = threadIdx.x & 31;
  int m0 = blockIdx.x * 64 + wave * 16;
  const __bf16* A0 = att  + (size_t)b * NPM * DD;
  const __bf16* A1 = feat + (size_t)b * NPM * DD;
  f32x8 acc[5] = {};
  for (int k = 0; k < KHEAD; k += 32) {
    const __bf16* Ab = (k < DD) ? A0 : A1;
    int kk = (k < DD) ? k : (k - DD);
    bf16x16 a = load_a_frag(Ab, DD, m0, kk, lane);
#pragma unroll
    for (int t = 0; t < 5; ++t) {
      bf16x16 bf = load_b_frag(W, KHEAD, 16 * t, k, lane);
      acc[t] = WMMA_BF16(a, bf, acc[t]);
    }
  }
#pragma unroll
  for (int t = 0; t < 5; ++t)
#pragma unroll
    for (int r = 0; r < 8; ++r) {
      int row = m0 + r + ((lane >> 4) << 3);
      int c   = 16 * t + (lane & 15);
      if (row < NN) {
        float v = acc[t][r];
        float* orow = out + ((size_t)b * NN + row) * NOUT;
        if (c < 2) {
          orow[c] = v + cls_b[c];
          if (c == 0) {  // piggyback the anchor passthrough columns
            orow[2] = anchors[(size_t)row * NOUT + 2];
            orow[3] = anchors[(size_t)row * NOUT + 3];
          }
        } else if (c < 75) {
          orow[c + 2] = v + reg_b[c - 2] + anchors[(size_t)row * NOUT + (c + 2)];
        }
      }
    }
}

// ---------------------------------------------------------------------------
extern "C" void kernel_launch(void* const* d_in, const int* in_sizes, int n_in,
                              void* d_out, int out_size, void* d_ws, size_t ws_size,
                              hipStream_t stream) {
  const float*   FV      = (const float*)d_in[0];
  const float*   attn_w  = (const float*)d_in[1];
  const float*   attn_b  = (const float*)d_in[2];
  const float*   cls_w   = (const float*)d_in[3];
  const float*   cls_b   = (const float*)d_in[4];
  const float*   reg_w   = (const float*)d_in[5];
  const float*   reg_b   = (const float*)d_in[6];
  const float*   anchors = (const float*)d_in[7];
  const int*     x_idx   = (const int*)d_in[10];
  const uint8_t* inval   = (const uint8_t*)d_in[11];
  float* out = (float*)d_out;

  // workspace layout (att overlays logits: logits are dead after softmax)
  char* ws = (char*)d_ws;
  size_t o = 0;
  __bf16* feat  = (__bf16*)(ws + o); o += (size_t)BB * NPM * DD * 2;     // 48 MB
  __bf16* featT = (__bf16*)(ws + o); o += (size_t)BB * DD * NPM * 2;     // 48 MB
  __bf16* Wt1   = (__bf16*)(ws + o); o += (size_t)NPC * DD * 2;          // 1.5 MB
  __bf16* hW    = (__bf16*)(ws + o); o += (size_t)HCOL * KHEAD * 2;      // 240 KB
  __bf16* S     = (__bf16*)(ws + o); o += (size_t)BB * NPM * NPC * 2;    // 64 MB
  float*  logits= (float*)(ws + o);                                      // 128 MB
  __bf16* att   = (__bf16*)(ws + o);                                     // overlay

  hipLaunchKernelGGL(lane_gather_kernel, dim3(NPM, BB), dim3(256), 0, stream,
                     FV, x_idx, inval, feat, featT);
  hipLaunchKernelGGL(lane_build_wt1_kernel, dim3(NPC), dim3(256), 0, stream,
                     attn_w, Wt1);
  hipLaunchKernelGGL(lane_build_hw_kernel, dim3(HCOL), dim3(256), 0, stream,
                     cls_w, reg_w, hW);
  hipLaunchKernelGGL(lane_gemm1_kernel, dim3(NPM / 128, NPC / 64, BB), dim3(128), 0, stream,
                     feat, Wt1, attn_b, logits);
  hipLaunchKernelGGL(lane_softmax_kernel, dim3(NPM, BB), dim3(256), 0, stream,
                     logits, S);
  hipLaunchKernelGGL(lane_gemm2_kernel, dim3(NPM / 128, DD / 64, BB), dim3(128), 0, stream,
                     S, featT, att);
  hipLaunchKernelGGL(lane_gemm3_kernel, dim3(NPM / 64, BB), dim3(128), 0, stream,
                     att, feat, hW, cls_b, reg_b, anchors, out);
}
